// SimpleSeqDecoder_74053826117791
// MI455X (gfx1250) — matrix operand order repaired and checked by reading, not compiled, not run on hardware
//
#include <hip/hip_runtime.h>
#include <hip/hip_bf16.h>
#include <math.h>

// ---------------------------------------------------------------------------
// SimpleSeqDecoder on MI455X (gfx1250, wave32, WMMA).
//
//  1) prep: zero LSTM state, combine biases; convert all weight matrices
//     fp32 -> bf16 (RNE) once into workspace.
//  2) K/V projections: two big bf16-WMMA GEMMs [32768,512]x[512,512]^T.
//  3) 31 sequential steps: q-proj GEMM, scalar attention, out-proj GEMM
//     (ctx -> X[:,512:]), LSTM gate GEMMs (accumulating), pointwise cells
//     (h2 -> X[:,:512]).
//  4) One deferred classifier GEMM [1984,1024]x[32000,1024]^T -> d_out with
//     the (t,b)->(b,t) row remap (mask/shift, b=2^6) folded into the store.
//
// GEMM inner loop: B fragments loaded directly as bf16 (2x b128 per tile),
// A fragments packed from fp32 with v_perm_b32 (1 op / 2 elems), 8 WMMAs
// per 32-K step (32M x 64N strip per wave).
// ---------------------------------------------------------------------------

typedef __attribute__((ext_vector_type(16))) __bf16 v16bf;
typedef __attribute__((ext_vector_type(8)))  float  v8f;

#define WAVES_PER_BLOCK 8

union AFrag { v16bf v; unsigned u[8]; };

// Pack 8 fp32 -> 8 bf16 (truncation) into 4 dwords using v_perm_b32.
static __device__ __forceinline__ void pack8(const float* __restrict__ p,
                                             unsigned* __restrict__ dst) {
    const uint4 lo = *(const uint4*)p;
    const uint4 hi = *(const uint4*)(p + 4);
    dst[0] = __builtin_amdgcn_perm(lo.y, lo.x, 0x07060302u);
    dst[1] = __builtin_amdgcn_perm(lo.w, lo.z, 0x07060302u);
    dst[2] = __builtin_amdgcn_perm(hi.y, hi.x, 0x07060302u);
    dst[3] = __builtin_amdgcn_perm(hi.w, hi.z, 0x07060302u);
}

// One-time fp32 -> bf16 (round-to-nearest-even) weight conversion.
__global__ void f32_to_bf16_kernel(const float* __restrict__ src,
                                   __bf16* __restrict__ dst, long n) {
    long i = (long)blockIdx.x * blockDim.x + threadIdx.x;
    const long stride = (long)gridDim.x * blockDim.x;
    for (; i < n; i += stride) {
        unsigned u = __builtin_bit_cast(unsigned, src[i]);
        u += 0x7FFFu + ((u >> 16) & 1u);
        unsigned short h = (unsigned short)(u >> 16);
        dst[i] = __builtin_bit_cast(__bf16, h);
    }
}

// C[M,N] = alpha*(opt_gather(A_f32)[M,K] @ B_bf16[N,K]^T + bias) (or +=).
// One wave computes a 32(M) x 64(N) strip: 2 A fragments x 4 B tiles,
// 8 WMMAs per 32-K step. Layouts per CDNA5 ISA 7.12.2 (wave32).
__global__ __launch_bounds__(256) void wmma_gemm_bf16(
    const float* __restrict__ A, int lda,
    const int* __restrict__ a_idx, int a_idx_stride,   // optional row gather
    const __bf16* __restrict__ Bw, int ldb,
    const float* __restrict__ bias, float alpha,
    float* __restrict__ C, int ldc, int accumulate,
    int M, int N, int K,
    int remap_lgb, int remap_t)   // out-row remap: m=t*2^lgb+b -> b*remap_t+t
{
    const int lane = threadIdx.x & 31;
    const int wave = threadIdx.x >> 5;
    const int mTiles  = M >> 5;   // 32-row strips
    const int nStrips = N >> 6;   // 64-col strips
    const int strip = blockIdx.x * WAVES_PER_BLOCK + wave;
    if (strip >= mTiles * nStrips) return;
    const int mt = strip % mTiles;
    const int ns = strip / mTiles;
    const int m0 = mt << 5;
    const int n0 = ns << 6;
    const int half = lane >> 4;      // 0: lanes 0-15, 1: lanes 16-31
    const int lm   = lane & 15;

    const int am0 = m0 + lm;
    const int am1 = m0 + 16 + lm;
    const long ar0 = a_idx ? (long)a_idx[(long)am0 * a_idx_stride] : (long)am0;
    const long ar1 = a_idx ? (long)a_idx[(long)am1 * a_idx_stride] : (long)am1;
    const float*  __restrict__ arow0 = A  + ar0 * (long)lda;
    const float*  __restrict__ arow1 = A  + ar1 * (long)lda;
    const __bf16* __restrict__ brow0 = Bw + (long)(n0 + lm) * (long)ldb;

    v8f acc0[4] = {};
    v8f acc1[4] = {};

    for (int k0 = 0; k0 < K; k0 += 32) {
        // A 16x32 bf16 fragment: elems 0..7 = K (8*half)+0..7,
        //                        elems 8..15 = K 16+(8*half)+0..7
        AFrag a0, a1;
        pack8(arow0 + k0 + 8 * half,      a0.u);
        pack8(arow0 + k0 + 16 + 8 * half, a0.u + 4);
        pack8(arow1 + k0 + 8 * half,      a1.u);
        pack8(arow1 + k0 + 16 + 8 * half, a1.u + 4);

#pragma unroll
        for (int j = 0; j < 4; ++j) {
            // B 32x16 fragment: column = lm, elems 0..7 = K (8*half)+0..7,
            // elems 8..15 = K 16+(8*half)+0..7 (two b128 loads).
            typedef __attribute__((ext_vector_type(8))) __bf16 v8bf;
            const __bf16* bp = brow0 + (long)j * 16 * ldb + k0 + 8 * half;
            const v8bf blo = *(const v8bf*)(bp);
            const v8bf bhi = *(const v8bf*)(bp + 16);
            const v16bf b = __builtin_shufflevector(blo, bhi, 0, 1, 2, 3, 4, 5, 6, 7, 8, 9, 10, 11, 12, 13, 14, 15);
            acc0[j] = __builtin_amdgcn_wmma_f32_16x16x32_bf16(
                false, a0.v, false, b, (short)0, acc0[j], false, false);
            acc1[j] = __builtin_amdgcn_wmma_f32_16x16x32_bf16(
                false, a1.v, false, b, (short)0, acc1[j], false, false);
        }
    }

    const int rmask = (1 << remap_lgb) - 1;   // remap_lgb==0 -> identity
    // C/D layout: VGPR r -> M = (strip base) + r + 8*half, N = n0 + j*16 + lm
#pragma unroll
    for (int j = 0; j < 4; ++j) {
        const int n = n0 + j * 16 + lm;
        const float bv = bias ? bias[n] : 0.0f;
#pragma unroll
        for (int r = 0; r < 8; ++r) {
            {
                const int m = m0 + r + 8 * half;
                const int orow = remap_lgb ? ((m & rmask) * remap_t + (m >> remap_lgb)) : m;
                float v = (acc0[j][r] + bv) * alpha;
                float* cp = C + (long)orow * ldc + n;
                if (accumulate) *cp += v; else *cp = v;
            }
            {
                const int m = m0 + 16 + r + 8 * half;
                const int orow = remap_lgb ? ((m & rmask) * remap_t + (m >> remap_lgb)) : m;
                float v = (acc1[j][r] + bv) * alpha;
                float* cp = C + (long)orow * ldc + n;
                if (accumulate) *cp += v; else *cp = v;
            }
        }
    }
}

// One workgroup per (batch, head): masked softmax attention over T=512, HD=64.
__global__ __launch_bounds__(256) void attention_kernel(
    const float* __restrict__ q,       // [64,512], already scaled by 1/sqrt(HD)
    const float* __restrict__ kp,      // [64,512,512]
    const float* __restrict__ vp,      // [64,512,512]
    const int*   __restrict__ lengths, // [64]
    float* __restrict__ ctx_raw)       // [64,512]
{
    __shared__ float qs[64];
    __shared__ float sc[512];
    __shared__ float red[256];
    __shared__ float part[4][64];
    const int b    = blockIdx.x >> 3;
    const int head = blockIdx.x & 7;
    const int tid  = threadIdx.x;
    const int len  = lengths[b];

    if (tid < 64) qs[tid] = q[b * 512 + head * 64 + tid];
    __syncthreads();

    const float* kb = kp + ((long)b * 512) * 512 + head * 64;
    for (int t = tid; t < 512; t += 256) {
        float s = -INFINITY;
        if (t < len) {
            const float* kr = kb + (long)t * 512;
            float accd = 0.f;
#pragma unroll
            for (int d = 0; d < 64; ++d) accd += qs[d] * kr[d];
            s = accd;
        }
        sc[t] = s;
    }
    __syncthreads();

    red[tid] = fmaxf(sc[tid], sc[tid + 256]);
    __syncthreads();
    for (int off = 128; off > 0; off >>= 1) {
        if (tid < off) red[tid] = fmaxf(red[tid], red[tid + off]);
        __syncthreads();
    }
    const float mx = red[0];
    __syncthreads();

    float e0 = __expf(sc[tid] - mx);
    float e1 = __expf(sc[tid + 256] - mx);
    sc[tid] = e0; sc[tid + 256] = e1;
    red[tid] = e0 + e1;
    __syncthreads();
    for (int off = 128; off > 0; off >>= 1) {
        if (tid < off) red[tid] += red[tid + off];
        __syncthreads();
    }
    const float inv = 1.0f / red[0];
    __syncthreads();

    const int d   = tid & 63;
    const int seg = tid >> 6;                 // 4 segments of 128 timesteps
    const float* vb = vp + ((long)b * 512) * 512 + head * 64 + d;
    float acc = 0.f;
    for (int t = seg * 128; t < seg * 128 + 128; ++t)
        acc += sc[t] * vb[(long)t * 512];
    part[seg][d] = acc;
    __syncthreads();
    if (tid < 64) {
        float r = (part[0][tid] + part[1][tid] + part[2][tid] + part[3][tid]) * inv;
        ctx_raw[b * 512 + head * 64 + tid] = r;
    }
}

// LSTM pointwise: gates g[64,2048] (i,f,g,o) + c -> h', c'. Optionally mirrors
// h' into the classifier input buffer X (row stride 1024).
__global__ __launch_bounds__(256) void lstm_cell_kernel(
    const float* __restrict__ g,
    const float* __restrict__ c_in,
    float* __restrict__ h_out,
    float* __restrict__ c_out,
    float* __restrict__ x_out)
{
    int idx = blockIdx.x * blockDim.x + threadIdx.x;
    if (idx >= 64 * 512) return;
    int b = idx >> 9, j = idx & 511;
    const float* gr = g + (long)b * 2048;
    float gi = gr[j], gf = gr[512 + j], gg = gr[1024 + j], go = gr[1536 + j];
    float si = 1.f / (1.f + __expf(-gi));
    float sf = 1.f / (1.f + __expf(-gf));
    float so = 1.f / (1.f + __expf(-go));
    float c2 = sf * c_in[idx] + si * tanhf(gg);
    float h2 = so * tanhf(c2);
    c_out[idx] = c2;
    h_out[idx] = h2;
    if (x_out) x_out[(long)b * 1024 + j] = h2;
}

// Zero LSTM state, precompute combined gate biases.
__global__ void prep_kernel(const float* __restrict__ b_ih1, const float* __restrict__ b_hh1,
                            const float* __restrict__ b_ih2, const float* __restrict__ b_hh2,
                            float* __restrict__ b1s, float* __restrict__ b2s,
                            float* __restrict__ states)
{
    int i = blockIdx.x * blockDim.x + threadIdx.x;
    if (i < 2048) { b1s[i] = b_ih1[i] + b_hh1[i]; b2s[i] = b_ih2[i] + b_hh2[i]; }
    if (i < 4 * 64 * 512) states[i] = 0.f;
}

extern "C" void kernel_launch(void* const* d_in, const int* in_sizes, int n_in,
                              void* d_out, int out_size, void* d_ws, size_t ws_size,
                              hipStream_t stream)
{
    (void)in_sizes; (void)n_in; (void)out_size; (void)ws_size;
    const float* keys       = (const float*)d_in[0];
    const float* values     = (const float*)d_in[1];
    const int*   lengths    = (const int*)  d_in[2];
    const int*   targets    = (const int*)  d_in[3];   // [64,32]
    const float* embedding  = (const float*)d_in[4];   // [32000,512]
    const float* w_ih1      = (const float*)d_in[5];   // [2048,1024]
    const float* b_ih1      = (const float*)d_in[6];
    const float* w_hh1      = (const float*)d_in[7];   // [2048,512]
    const float* b_hh1      = (const float*)d_in[8];
    const float* w_ih2      = (const float*)d_in[9];   // [2048,512]
    const float* b_ih2      = (const float*)d_in[10];
    const float* w_hh2      = (const float*)d_in[11];  // [2048,512]
    const float* b_hh2      = (const float*)d_in[12];
    const float* in_proj_w  = (const float*)d_in[13];  // [1536,512] = wq|wk|wv
    const float* in_proj_b  = (const float*)d_in[14];
    const float* out_proj_w = (const float*)d_in[15];  // [512,512]
    const float* out_proj_b = (const float*)d_in[16];
    const float* cls_w      = (const float*)d_in[17];  // [32000,1024]
    const float* cls_b      = (const float*)d_in[18];
    float* out = (float*)d_out;                        // [64,31,32000]

    // workspace carve-up (256B aligned)
    char* ws = (char*)d_ws;
    size_t off = 0;
    auto allocB = [&](size_t nbytes) -> char* {
        char* p = ws + off;
        off += ((nbytes + 255) & ~(size_t)255);
        return p;
    };
    auto allocF  = [&](size_t n) -> float*  { return (float*) allocB(n * 4); };
    auto allocBF = [&](size_t n) -> __bf16* { return (__bf16*)allocB(n * 2); };

    float* k_proj = allocF(64L * 512 * 512);   // 64 MB
    float* v_proj = allocF(64L * 512 * 512);   // 64 MB
    float* X      = allocF(1984L * 1024);      // [t*64+b][ h2 | ctx ]
    float* qb     = allocF(64 * 512);
    float* ctxr   = allocF(64 * 512);
    float* gbuf   = allocF(64 * 2048);
    float* states = allocF(4L * 64 * 512);
    float* h1 = states, *c1 = states + 64 * 512;
    float* h2 = states + 2 * 64 * 512, *c2 = states + 3 * 64 * 512;
    float* b1s = allocF(2048);
    float* b2s = allocF(2048);
    // bf16 weight copies
    __bf16* cls_w_bf   = allocBF(32000L * 1024);
    __bf16* w_ih1_bf   = allocBF(2048L * 1024);
    __bf16* w_hh1_bf   = allocBF(2048L * 512);
    __bf16* w_ih2_bf   = allocBF(2048L * 512);
    __bf16* w_hh2_bf   = allocBF(2048L * 512);
    __bf16* in_proj_bf = allocBF(1536L * 512);
    __bf16* out_proj_bf= allocBF(512L * 512);

    auto cvt = [&](const float* src, __bf16* dst, long n) {
        int blocks = (int)((n + 255) / 256);
        if (blocks > 4096) blocks = 4096;
        f32_to_bf16_kernel<<<dim3(blocks), dim3(256), 0, stream>>>(src, dst, n);
    };
    auto gemm = [&](const float* A, int lda, const int* aidx, int aistride,
                    const __bf16* Bw, int ldb, const float* bias, float alpha,
                    float* C, int ldc, int accum, int M, int N, int K,
                    int rlgb, int rt) {
        int strips = (M / 32) * (N / 64);
        int blocks = (strips + WAVES_PER_BLOCK - 1) / WAVES_PER_BLOCK;
        wmma_gemm_bf16<<<dim3(blocks), dim3(256), 0, stream>>>(
            A, lda, aidx, aistride, Bw, ldb, bias, alpha,
            C, ldc, accum, M, N, K, rlgb, rt);
    };

    prep_kernel<<<dim3(512), dim3(256), 0, stream>>>(
        b_ih1, b_hh1, b_ih2, b_hh2, b1s, b2s, states);

    // one-time weight conversions (RNE)
    cvt(cls_w,      cls_w_bf,    32000L * 1024);
    cvt(w_ih1,      w_ih1_bf,    2048L * 1024);
    cvt(w_hh1,      w_hh1_bf,    2048L * 512);
    cvt(w_ih2,      w_ih2_bf,    2048L * 512);
    cvt(w_hh2,      w_hh2_bf,    2048L * 512);
    cvt(in_proj_w,  in_proj_bf,  1536L * 512);
    cvt(out_proj_w, out_proj_bf, 512L * 512);

    // One-time K/V projections: [B*T,512] @ wk/wv^T + bias
    gemm(keys,   512, nullptr, 0, in_proj_bf + 512L  * 512, 512, in_proj_b + 512,  1.0f,
         k_proj, 512, 0, 64 * 512, 512, 512, 0, 0);
    gemm(values, 512, nullptr, 0, in_proj_bf + 1024L * 512, 512, in_proj_b + 1024, 1.0f,
         v_proj, 512, 0, 64 * 512, 512, 512, 0, 0);

    const float scale = 0.125f;  // 1/sqrt(HD=64)
    for (int t = 0; t < 31; ++t) {
        float* xrow = X + (size_t)t * 64 * 1024;
        // q = (h2 @ wq^T + bq) * scale
        gemm(h2, 512, nullptr, 0, in_proj_bf, 512, in_proj_b, scale,
             qb, 512, 0, 64, 512, 512, 0, 0);
        // masked softmax attention -> ctx_raw
        attention_kernel<<<dim3(512), dim3(256), 0, stream>>>(
            qb, k_proj, v_proj, lengths, ctxr);
        // ctx = ctx_raw @ out_proj^T + b  -> X[:, 512:1024]
        gemm(ctxr, 512, nullptr, 0, out_proj_bf, 512, out_proj_b, 1.0f,
             xrow + 512, 1024, 0, 64, 512, 512, 0, 0);
        // g1 = embed[targets[:,t]] @ w_ih1[:, :512]^T + (b_ih1 + b_hh1)
        gemm(embedding, 512, targets + t, 32, w_ih1_bf, 1024, b1s, 1.0f,
             gbuf, 2048, 0, 64, 2048, 512, 0, 0);
        // g1 += ctx @ w_ih1[:, 512:]^T
        gemm(xrow + 512, 1024, nullptr, 0, w_ih1_bf + 512, 1024, nullptr, 1.0f,
             gbuf, 2048, 1, 64, 2048, 512, 0, 0);
        // g1 += h1 @ w_hh1^T
        gemm(h1, 512, nullptr, 0, w_hh1_bf, 512, nullptr, 1.0f,
             gbuf, 2048, 1, 64, 2048, 512, 0, 0);
        lstm_cell_kernel<<<dim3(128), dim3(256), 0, stream>>>(
            gbuf, c1, h1, c1, (float*)nullptr);
        // g2 = h1 @ w_ih2^T + (b_ih2 + b_hh2); g2 += h2_old @ w_hh2^T
        gemm(h1, 512, nullptr, 0, w_ih2_bf, 512, b2s, 1.0f,
             gbuf, 2048, 0, 64, 2048, 512, 0, 0);
        gemm(h2, 512, nullptr, 0, w_hh2_bf, 512, nullptr, 1.0f,
             gbuf, 2048, 1, 64, 2048, 512, 0, 0);
        lstm_cell_kernel<<<dim3(128), dim3(256), 0, stream>>>(
            gbuf, c2, h2, c2, xrow);   // h2 -> X[:, :512]
    }

    // Deferred classifier: out[b*31+t, :] = X[t*64+b, :] @ cls_w^T + cls_b
    // remap_lgb = 6 (B=64), remap_t = 31
    gemm(X, 1024, nullptr, 0, cls_w_bf, 1024, cls_b, 1.0f,
         out, 32000, 0, 1984, 32000, 1024, 6, 31);
}
